// AudioTransformer_41051297415679
// MI455X (gfx1250) — matrix operand order repaired and checked by reading, not compiled, not run on hardware
//
#include <hip/hip_runtime.h>
#include <hip/hip_bf16.h>

typedef __attribute__((ext_vector_type(16))) _Float16 v16h;
typedef __attribute__((ext_vector_type(8)))  _Float16 v8h;
typedef __attribute__((ext_vector_type(8)))  float    v8f;
typedef __attribute__((ext_vector_type(4)))  int      v4i;

#define S_LEN   1024
#define D_DIM   256
#define FF_DIM  1024
#define SP_LEN  1312   /* 128 pad + 1024 + 160 pad */
#define N_BATCH 2

#if defined(__has_builtin)
#  if __has_builtin(__builtin_amdgcn_global_load_async_to_lds_b128)
#    define HAVE_ASYNC 1
#  endif
#endif
#ifndef HAVE_ASYNC
#  define HAVE_ASYNC 0
#endif

__device__ __forceinline__ v8f wmma16x16x32(v16h a, v16h b, v8f c) {
  // D = A(16x32 f16) * B(32x16 f16) + C(16x16 f32)
  return __builtin_amdgcn_wmma_f32_16x16x32_f16(false, a, false, b, (short)0, c,
                                                false, false);
}

#if HAVE_ASYNC
typedef __attribute__((address_space(1))) v4i* gv4_p;  // global <4 x i32>*
typedef __attribute__((address_space(3))) v4i* lv4_p;  // LDS    <4 x i32>*
__device__ __forceinline__ void async_cp16(const void* g, void* l) {
  __builtin_amdgcn_global_load_async_to_lds_b128((gv4_p)(void*)g, (lv4_p)l, 0, 0);
}
__device__ __forceinline__ void wait_async0() {
#if __has_builtin(__builtin_amdgcn_s_wait_asynccnt)
  __builtin_amdgcn_s_wait_asynccnt(0);
#else
  asm volatile("s_wait_asynccnt 0" ::: "memory");
#endif
}
#endif

// ---------------------------------------------------------------------------
// GEMM: C[M,N] = A[M,K](f16) * Bt[N,K](f16, pre-transposed weights) + bias
// 256 threads = 8 waves; 128x64 C tile; K stepped by 32; double-buffered LDS
// filled via async-load-to-LDS when available.
// Epilogue modes: 0 -> f32 [M][N]; 1 -> f16 [M][N];
//                 2 -> f16 seq-padded K layout [B][SP][N] (+128 row offset);
//                 3 -> f16 transposed V layout [B][N][SP] (+128 col offset).
// ---------------------------------------------------------------------------
__global__ __launch_bounds__(256) void gemm_f16_kernel(
    const _Float16* __restrict__ A, const _Float16* __restrict__ Bt,
    const float* __restrict__ bias, float* __restrict__ Cf,
    _Float16* __restrict__ Ch, int M, int N, int K, int relu, int mode)
{
  __shared__ alignas(16) _Float16 sA[2][128][32];
  __shared__ alignas(16) _Float16 sB[2][64][32];
  const int tid  = threadIdx.x;
  const int wave = tid >> 5;
  const int lane = tid & 31;
  const int l16  = lane & 15;
  const int g    = lane >> 4;
  const int bm   = blockIdx.x * 128;
  const int bn   = blockIdx.y * 64;

  v8f acc[4] = {};

  auto stage = [&](int buf, int k0) {
#if HAVE_ASYNC
    // A tile: 128x32 halves = 512 16B chunks; B tile: 64x32 = 256 chunks.
    {
      int c0 = tid, c1 = tid + 256;
      async_cp16(&A[(size_t)(bm + (c0 >> 2)) * K + k0 + ((c0 & 3) << 3)],
                 &sA[buf][c0 >> 2][(c0 & 3) << 3]);
      async_cp16(&A[(size_t)(bm + (c1 >> 2)) * K + k0 + ((c1 & 3) << 3)],
                 &sA[buf][c1 >> 2][(c1 & 3) << 3]);
      async_cp16(&Bt[(size_t)(bn + (tid >> 2)) * K + k0 + ((tid & 3) << 3)],
                 &sB[buf][tid >> 2][(tid & 3) << 3]);
    }
#else
    {
      int rA = tid >> 1, cA = (tid & 1) << 4;  // 2 threads/row, 16 halves each
      *(v8h*)&sA[buf][rA][cA]     = *(const v8h*)&A[(size_t)(bm + rA) * K + k0 + cA];
      *(v8h*)&sA[buf][rA][cA + 8] = *(const v8h*)&A[(size_t)(bm + rA) * K + k0 + cA + 8];
      int rB = tid >> 2, cB = (tid & 3) << 3;
      *(v8h*)&sB[buf][rB][cB] = *(const v8h*)&Bt[(size_t)(bn + rB) * K + k0 + cB];
    }
#endif
  };

  const int nk = K >> 5;
  stage(0, 0);
#if HAVE_ASYNC
  wait_async0();
#endif
  __syncthreads();

  for (int i = 0; i < nk; ++i) {
    int cur = i & 1;
    if (i + 1 < nk) stage(cur ^ 1, (i + 1) << 5);

    // A fragment: lanes 0-15 hold K {0..15}, 16-31 {16..31} (matches B)
    union { v16h v; v8h h[2]; } a;
    a.h[0] = *(const v8h*)&sA[cur][wave * 16 + l16][16 * g];
    a.h[1] = *(const v8h*)&sA[cur][wave * 16 + l16][16 * g + 8];
#pragma unroll
    for (int t = 0; t < 4; ++t) {
      union { v16h v; v8h h[2]; } b;
      b.h[0] = *(const v8h*)&sB[cur][t * 16 + l16][16 * g];
      b.h[1] = *(const v8h*)&sB[cur][t * 16 + l16][16 * g + 8];
      acc[t] = wmma16x16x32(a.v, b.v, acc[t]);
    }
#if HAVE_ASYNC
    if (i + 1 < nk) wait_async0();
#endif
    __syncthreads();
  }

#pragma unroll
  for (int t = 0; t < 4; ++t) {
    int col = bn + t * 16 + l16;
    float bv = bias[col];
#pragma unroll
    for (int v = 0; v < 8; ++v) {
      int row = bm + wave * 16 + v + 8 * g;    // C layout: elem v -> row v+8g
      float val = acc[t][v] + bv;
      if (relu) val = fmaxf(val, 0.f);
      if (mode == 0) {
        Cf[(size_t)row * N + col] = val;
      } else if (mode == 1) {
        Ch[(size_t)row * N + col] = (_Float16)val;
      } else if (mode == 2) {
        int bb = row / S_LEN, s = row % S_LEN;
        Ch[((size_t)bb * SP_LEN + 128 + s) * N + col] = (_Float16)val;
      } else {
        int bb = row / S_LEN, s = row % S_LEN;
        Ch[((size_t)bb * D_DIM + col) * SP_LEN + 128 + s] = (_Float16)val;
      }
    }
  }
}

// ---------------------------------------------------------------------------
// Local attention: one 16-query tile per workgroup (128 thr = 4 waves).
// Energies vs 272-col key window via WMMA, softmax (structural window mask
// only; sequence OOB handled by zero padding == reference semantics),
// then attn @ V via WMMA.
// ---------------------------------------------------------------------------
__global__ __launch_bounds__(128) void local_attn_kernel(
    const _Float16* __restrict__ qh,   // [B][S][D] f16
    const _Float16* __restrict__ khp,  // [B][SP][D] f16, seq offset +128, zero-padded
    const _Float16* __restrict__ vTp,  // [B][D][SP] f16, zero-padded
    float* __restrict__ out)           // [B][S][D] f32
{
  __shared__ alignas(16) float    sE[16][288];
  __shared__ alignas(16) _Float16 sQ[16][256];
  __shared__ alignas(16) _Float16 sAttn[16][288];
  __shared__ float sRed[16][8];

  const int b    = blockIdx.y;
  const int s0   = blockIdx.x * 16;
  const int tid  = threadIdx.x;
  const int wave = tid >> 5;
  const int lane = tid & 31;
  const int l16  = lane & 15;
  const int g    = lane >> 4;

  // Stage query tile (16x256 f16) to LDS
  for (int i = tid; i < 512; i += 128) {
    int r = i >> 5, c = (i & 31) << 3;
    *(v8h*)&sQ[r][c] = *(const v8h*)&qh[((size_t)(b * S_LEN + s0 + r)) * D_DIM + c];
  }
  __syncthreads();

  // Phase 1: energies, 17 key tiles distributed across 4 waves
  const _Float16* kbase = khp + ((size_t)b * SP_LEN + s0) * D_DIM; // row j -> abs key s0-128+j
  for (int tile = wave; tile < 17; tile += 4) {
    v8f acc = {};
#pragma unroll
    for (int kk = 0; kk < 8; ++kk) {
      union { v16h v; v8h h[2]; } a, bb;
      a.h[0] = *(const v8h*)&sQ[l16][kk * 32 + 16 * g];
      a.h[1] = *(const v8h*)&sQ[l16][kk * 32 + 16 * g + 8];
      const _Float16* kr = kbase + (size_t)(tile * 16 + l16) * D_DIM + kk * 32 + 16 * g;
      bb.h[0] = *(const v8h*)kr;
      bb.h[1] = *(const v8h*)(kr + 8);
      acc = wmma16x16x32(a.v, bb.v, acc);
    }
#pragma unroll
    for (int v = 0; v < 8; ++v) sE[v + 8 * g][tile * 16 + l16] = acc[v];
  }
  __syncthreads();

  // Phase 2: softmax. thread -> (row r, chunk ch of 34 cols); window mask r<=j<=r+256.
  {
    int r = tid & 15, ch = tid >> 4;
    int j0 = ch * 34, j1 = j0 + 34;
    float mx = -3.0e38f;
    for (int j = j0; j < j1; ++j)
      if (j >= r && j <= r + 256) mx = fmaxf(mx, sE[r][j]);
    sRed[r][ch] = mx;
    __syncthreads();
    float rowmax = sRed[r][0];
#pragma unroll
    for (int i = 1; i < 8; ++i) rowmax = fmaxf(rowmax, sRed[r][i]);
    __syncthreads();
    float sum = 0.f;
    for (int j = j0; j < j1; ++j) {
      float e = (j >= r && j <= r + 256) ? __expf(sE[r][j] - rowmax) : 0.f;
      sE[r][j] = e;
      sum += e;
    }
    sRed[r][ch] = sum;
    __syncthreads();
    float tot = 0.f;
#pragma unroll
    for (int i = 0; i < 8; ++i) tot += sRed[r][i];
    float inv = 1.f / tot;
    for (int j = j0; j < j1; ++j) sAttn[r][j] = (_Float16)(sE[r][j] * inv);
    if (ch == 0)
      for (int j = 272; j < 288; ++j) sAttn[r][j] = (_Float16)0.f;
  }
  __syncthreads();

  // Phase 3: out = attn(16x288) @ V; each wave owns 64 output dims
  const _Float16* vbase = vTp + (size_t)b * D_DIM * SP_LEN;
  v8f acc2[4] = {};
  for (int kk = 0; kk < 9; ++kk) {
    union { v16h v; v8h h[2]; } a;
    a.h[0] = *(const v8h*)&sAttn[l16][kk * 32 + 16 * g];
    a.h[1] = *(const v8h*)&sAttn[l16][kk * 32 + 16 * g + 8];
#pragma unroll
    for (int t = 0; t < 4; ++t) {
      int d = wave * 64 + t * 16 + l16;
      const _Float16* vr = vbase + (size_t)d * SP_LEN + s0 + kk * 32 + 16 * g;
      union { v16h v; v8h h[2]; } bb;
      bb.h[0] = *(const v8h*)vr;
      bb.h[1] = *(const v8h*)(vr + 8);
      acc2[t] = wmma16x16x32(a.v, bb.v, acc2[t]);
    }
  }
#pragma unroll
  for (int t = 0; t < 4; ++t)
#pragma unroll
    for (int v = 0; v < 8; ++v)
      out[((size_t)(b * S_LEN + s0 + v + 8 * g)) * D_DIM + wave * 64 + t * 16 + l16] =
          acc2[t][v];
}

// ---------------------------------------------------------------------------
// x = LayerNorm(a + x) * gamma + beta; also emits f16 copy for next GEMM.
// One wave32 per row of D=256.
// ---------------------------------------------------------------------------
__global__ __launch_bounds__(256) void add_ln_kernel(
    const float* __restrict__ a, float* __restrict__ x, _Float16* __restrict__ xh,
    const float* __restrict__ gamma, const float* __restrict__ beta)
{
  int row  = blockIdx.x * 8 + (threadIdx.x >> 5);
  int lane = threadIdx.x & 31;
  const float* ar = a + (size_t)row * D_DIM;
  float* xr = x + (size_t)row * D_DIM;
  _Float16* hr = xh + (size_t)row * D_DIM;
  float vals[8];
  float s = 0.f;
#pragma unroll
  for (int i = 0; i < 8; ++i) {
    float t = ar[lane + 32 * i] + xr[lane + 32 * i];
    vals[i] = t; s += t;
  }
#pragma unroll
  for (int off = 16; off > 0; off >>= 1) s += __shfl_xor(s, off, 32);
  float mean = s * (1.f / 256.f);
  float vs = 0.f;
#pragma unroll
  for (int i = 0; i < 8; ++i) { float d = vals[i] - mean; vs += d * d; }
#pragma unroll
  for (int off = 16; off > 0; off >>= 1) vs += __shfl_xor(vs, off, 32);
  float rstd = rsqrtf(vs * (1.f / 256.f) + 1e-5f);
#pragma unroll
  for (int i = 0; i < 8; ++i) {
    int col = lane + 32 * i;
    float o = (vals[i] - mean) * rstd * gamma[col] + beta[col];
    xr[col] = o;
    hr[col] = (_Float16)o;
  }
}

// ---------------------------------------------------------------------------
// conv1d(k=5, same) + bias + relu + maxpool2, NCH layout
// ---------------------------------------------------------------------------
__global__ void conv_relu_pool_kernel(
    const float* __restrict__ x, const float* __restrict__ w,
    const float* __restrict__ bias, float* __restrict__ y,
    int Cin, int Cout, int Lin)
{
  int idx = blockIdx.x * blockDim.x + threadIdx.x;
  int Lout = Lin >> 1;
  if (idx >= N_BATCH * Cout * Lout) return;
  int p  = idx % Lout;
  int co = (idx / Lout) % Cout;
  int b  = idx / (Lout * Cout);
  float acc0 = bias[co], acc1 = bias[co];
  const float* wr = w + (size_t)co * Cin * 5;
  const float* xb = x + (size_t)b * Cin * Lin;
  int i0 = 2 * p;
  for (int ci = 0; ci < Cin; ++ci) {
    const float* xr = xb + (size_t)ci * Lin;
    const float* wc = wr + ci * 5;
#pragma unroll
    for (int k = 0; k < 5; ++k) {
      int j0 = i0 + k - 2;
      float wv = wc[k];
      float x0 = (j0 >= 0 && j0 < Lin) ? xr[j0] : 0.f;
      int j1 = j0 + 1;
      float x1 = (j1 >= 0 && j1 < Lin) ? xr[j1] : 0.f;
      acc0 += wv * x0;
      acc1 += wv * x1;
    }
  }
  y[idx] = fmaxf(fmaxf(acc0, 0.f), fmaxf(acc1, 0.f));
}

// ---------------------------------------------------------------------------
// Transposed conv1d, stride 2, k=4, pad per reference: exactly 2 taps/output.
// w layout (Cin, Cout, 4). Lout = 2*Lin.
// ---------------------------------------------------------------------------
__global__ void deconv_kernel(
    const float* __restrict__ x, const float* __restrict__ w,
    const float* __restrict__ bias, float* __restrict__ y,
    int Cin, int Cout, int Lin, int relu)
{
  int idx = blockIdx.x * blockDim.x + threadIdx.x;
  int Lout = Lin * 2;
  if (idx >= N_BATCH * Cout * Lout) return;
  int o  = idx % Lout;
  int co = (idx / Lout) % Cout;
  int b  = idx / (Lout * Cout);
  int ia, ka, ib, kb;
  if ((o & 1) == 0) { ia = (o - 2) >> 1; ka = 3; ib = o >> 1;       kb = 1; }
  else              { ia = (o - 1) >> 1; ka = 2; ib = (o + 1) >> 1; kb = 0; }
  float acc = bias[co];
  const float* xb = x + (size_t)b * Cin * Lin;
  for (int ci = 0; ci < Cin; ++ci) {
    const float* xr = xb + (size_t)ci * Lin;
    const float* wr = w + ((size_t)ci * Cout + co) * 4;
    if (ia >= 0 && ia < Lin) acc += wr[ka] * xr[ia];
    if (ib >= 0 && ib < Lin) acc += wr[kb] * xr[ib];
  }
  if (relu) acc = fmaxf(acc, 0.f);
  y[idx] = acc;
}

// ---------------------------------------------------------------------------
// Small utility kernels
// ---------------------------------------------------------------------------
__global__ void wcvt_kernel(const float* __restrict__ w, _Float16* __restrict__ wT,
                            int K, int N) {
  int i = blockIdx.x * blockDim.x + threadIdx.x;  // i indexes wT[n][k]
  if (i >= K * N) return;
  int k = i % K, n = i / K;
  wT[i] = (_Float16)w[(size_t)k * N + n];
}
__global__ void zero_f16_kernel(_Float16* __restrict__ dst, int n8) {
  int i = blockIdx.x * blockDim.x + threadIdx.x;
  if (i < n8) { v8h z = {}; *(v8h*)&dst[(size_t)i * 8] = z; }
}
__global__ void cvt_rows_kernel(const float* __restrict__ src,
                                _Float16* __restrict__ dst, int n) {
  int i = blockIdx.x * blockDim.x + threadIdx.x;
  if (i < n) dst[i] = (_Float16)src[i];
}
__global__ void nch_to_nhc_kernel(const float* __restrict__ in,
                                  float* __restrict__ outp, int C, int Sl) {
  int i = blockIdx.x * blockDim.x + threadIdx.x;  // output idx over B*S*C
  if (i >= N_BATCH * C * Sl) return;
  int c = i % C;
  int s = (i / C) % Sl;
  int b = i / (C * Sl);
  outp[i] = in[((size_t)b * C + c) * Sl + s];
}
__global__ void nhc_to_nch_kernel(const float* __restrict__ in,
                                  float* __restrict__ outp, int C, int Sl) {
  int i = blockIdx.x * blockDim.x + threadIdx.x;  // output idx over B*C*S
  if (i >= N_BATCH * C * Sl) return;
  int s = i % Sl;
  int c = (i / Sl) % C;
  int b = i / (C * Sl);
  outp[i] = in[((size_t)b * Sl + s) * C + c];
}

// ---------------------------------------------------------------------------
extern "C" void kernel_launch(void* const* d_in, const int* in_sizes, int n_in,
                              void* d_out, int out_size, void* d_ws, size_t ws_size,
                              hipStream_t stream) {
  (void)in_sizes; (void)n_in; (void)out_size; (void)ws_size;
  const float* xin = (const float*)d_in[0];
  const float* encw[3] = {(const float*)d_in[1], (const float*)d_in[3], (const float*)d_in[5]};
  const float* encb[3] = {(const float*)d_in[2], (const float*)d_in[4], (const float*)d_in[6]};
  struct Lp { const float *wq,*bq,*wk,*bk,*wv,*bv,*w1,*b1,*w2,*b2,*g1,*be1,*g2,*be2; } L[4];
  for (int l = 0; l < 4; ++l) {
    int base = 7 + 14 * l;
    L[l].wq = (const float*)d_in[base+0];  L[l].bq  = (const float*)d_in[base+1];
    L[l].wk = (const float*)d_in[base+2];  L[l].bk  = (const float*)d_in[base+3];
    L[l].wv = (const float*)d_in[base+4];  L[l].bv  = (const float*)d_in[base+5];
    L[l].w1 = (const float*)d_in[base+6];  L[l].b1  = (const float*)d_in[base+7];
    L[l].w2 = (const float*)d_in[base+8];  L[l].b2  = (const float*)d_in[base+9];
    L[l].g1 = (const float*)d_in[base+10]; L[l].be1 = (const float*)d_in[base+11];
    L[l].g2 = (const float*)d_in[base+12]; L[l].be2 = (const float*)d_in[base+13];
  }
  const float* decw[3] = {(const float*)d_in[63], (const float*)d_in[65], (const float*)d_in[67]};
  const float* decb[3] = {(const float*)d_in[64], (const float*)d_in[66], (const float*)d_in[68]};

  char* ws = (char*)d_ws;
  size_t off = 0;
  auto alloc = [&](size_t bytes) { size_t r = off; off += (bytes + 255) & ~(size_t)255; return r; };

  _Float16 *wqT[4], *wkT[4], *wvT[4], *w1T[4], *w2T[4];
  for (int l = 0; l < 4; ++l) {
    wqT[l] = (_Float16*)(ws + alloc((size_t)256 * 256 * 2));
    wkT[l] = (_Float16*)(ws + alloc((size_t)256 * 256 * 2));
    wvT[l] = (_Float16*)(ws + alloc((size_t)256 * 256 * 2));
    w1T[l] = (_Float16*)(ws + alloc((size_t)1024 * 256 * 2));
    w2T[l] = (_Float16*)(ws + alloc((size_t)256 * 1024 * 2));
  }
  float*    bufH = (float*)(ws + alloc((size_t)2048 * 1024 * 4));   // 8 MB: enc1 / dec2; front 4MB = f16 FF h
  float*    bufA = (float*)(ws + alloc((size_t)2 * 256 * 2048 * 4));// 4 MB: enc2 / attn / dec1
  float*    bufX = (float*)(ws + alloc((size_t)2048 * 256 * 4));    // 2 MB: main x (f32)
  float*    bufT = (float*)(ws + alloc((size_t)2048 * 256 * 4));    // 2 MB: enc3 / y / xc
  _Float16* xh   = (_Float16*)(ws + alloc((size_t)2048 * 256 * 2)); // f16 activations
  _Float16* qh   = (_Float16*)(ws + alloc((size_t)2048 * 256 * 2));
  _Float16* khp  = (_Float16*)(ws + alloc((size_t)N_BATCH * SP_LEN * 256 * 2));
  _Float16* vTp  = (_Float16*)(ws + alloc((size_t)N_BATCH * 256 * SP_LEN * 2));
  _Float16* hh   = (_Float16*)bufH;                                 // f16 FF hidden, aliases bufH

  const int TB = 256;
  auto blocks = [](int total, int tb) { return (total + tb - 1) / tb; };

  // Weight convert+transpose (fp32 -> f16 [N][K])
  for (int l = 0; l < 4; ++l) {
    wcvt_kernel<<<blocks(256 * 256, TB), TB, 0, stream>>>(L[l].wq, wqT[l], 256, 256);
    wcvt_kernel<<<blocks(256 * 256, TB), TB, 0, stream>>>(L[l].wk, wkT[l], 256, 256);
    wcvt_kernel<<<blocks(256 * 256, TB), TB, 0, stream>>>(L[l].wv, wvT[l], 256, 256);
    wcvt_kernel<<<blocks(256 * 1024, TB), TB, 0, stream>>>(L[l].w1, w1T[l], 256, 1024);
    wcvt_kernel<<<blocks(1024 * 256, TB), TB, 0, stream>>>(L[l].w2, w2T[l], 1024, 256);
  }
  // Zero padded f16 K/V buffers (for reference-exact zero-pad attention)
  {
    int n8 = (N_BATCH * SP_LEN * 256) / 8;
    zero_f16_kernel<<<blocks(n8, TB), TB, 0, stream>>>(khp, n8);
    zero_f16_kernel<<<blocks(n8, TB), TB, 0, stream>>>(vTp, n8);
  }

  // Encoder
  conv_relu_pool_kernel<<<blocks(2 * 128 * 4096, TB), TB, 0, stream>>>(xin,  encw[0], encb[0], bufH, 80, 128, 8192);
  conv_relu_pool_kernel<<<blocks(2 * 192 * 2048, TB), TB, 0, stream>>>(bufH, encw[1], encb[1], bufA, 128, 192, 4096);
  conv_relu_pool_kernel<<<blocks(2 * 256 * 1024, TB), TB, 0, stream>>>(bufA, encw[2], encb[2], bufT, 192, 256, 2048);
  nch_to_nhc_kernel<<<blocks(2 * 256 * 1024, TB), TB, 0, stream>>>(bufT, bufX, 256, 1024);
  cvt_rows_kernel<<<blocks(2048 * 256, TB), TB, 0, stream>>>(bufX, xh, 2048 * 256);

  // Transformer layers (all matmuls on WMMA, f16 in / f32 accumulate)
  for (int l = 0; l < 4; ++l) {
    gemm_f16_kernel<<<dim3(16, 4), 256, 0, stream>>>(xh, wqT[l], L[l].bq, nullptr, qh,  2048, 256, 256, 0, 1);
    gemm_f16_kernel<<<dim3(16, 4), 256, 0, stream>>>(xh, wkT[l], L[l].bk, nullptr, khp, 2048, 256, 256, 0, 2);
    gemm_f16_kernel<<<dim3(16, 4), 256, 0, stream>>>(xh, wvT[l], L[l].bv, nullptr, vTp, 2048, 256, 256, 0, 3);

    local_attn_kernel<<<dim3(64, 2), 128, 0, stream>>>(qh, khp, vTp, bufA);
    add_ln_kernel<<<256, 256, 0, stream>>>(bufA, bufX, xh, L[l].g1, L[l].be1);

    gemm_f16_kernel<<<dim3(16, 16), 256, 0, stream>>>(xh, w1T[l], L[l].b1, nullptr, hh, 2048, 1024, 256, 1, 1);
    gemm_f16_kernel<<<dim3(16, 4), 256, 0, stream>>>(hh, w2T[l], L[l].b2, bufT, nullptr, 2048, 256, 1024, 0, 0);
    add_ln_kernel<<<256, 256, 0, stream>>>(bufT, bufX, xh, L[l].g2, L[l].be2);
  }

  // Decoder
  nhc_to_nch_kernel<<<blocks(2 * 256 * 1024, TB), TB, 0, stream>>>(bufX, bufT, 256, 1024);
  deconv_kernel<<<blocks(2 * 256 * 2048, TB), TB, 0, stream>>>(bufT, decw[0], decb[0], bufA, 256, 256, 1024, 1);
  deconv_kernel<<<blocks(2 * 256 * 4096, TB), TB, 0, stream>>>(bufA, decw[1], decb[1], bufH, 256, 256, 2048, 1);
  deconv_kernel<<<blocks(2 * 1 * 8192, TB), TB, 0, stream>>>(bufH, decw[2], decb[2], (float*)d_out, 256, 1, 4096, 0);
}
